// RNNSeq2Seq_8108898255312
// MI455X (gfx1250) — matrix operand order repaired and checked by reading, not compiled, not run on hardware
//
#include <hip/hip_runtime.h>
#include <hip/hip_bf16.h>

// ---------------- problem constants (match reference) ----------------
constexpr int B_   = 32;
constexpr int S_   = 64;
constexpr int T_   = 64;
constexpr int E_   = 256;
constexpr int H_   = 512;
constexpr int VT_  = 32000;
constexpr int G_   = 3 * H_;      // 1536 gates
constexpr int Tm1_ = T_ - 1;      // 63
constexpr int CH_  = 2 * H_;      // 1024 concat(d1, ctx)

// ---------------- types ----------------
typedef __bf16 bf16_t;
typedef __attribute__((ext_vector_type(16))) __bf16 v16bf;
typedef __attribute__((ext_vector_type(8)))  float  v8f;
typedef __attribute__((ext_vector_type(4)))  int    v4i;

struct __attribute__((aligned(16))) U4 { unsigned x, y, z, w; };

__device__ __forceinline__ bf16_t f2bf(float f) {
  unsigned u = __builtin_bit_cast(unsigned, f);
  unsigned r = u + 0x7FFFu + ((u >> 16) & 1u);   // round-to-nearest-even
  unsigned short h = (unsigned short)(r >> 16);
  return __builtin_bit_cast(bf16_t, h);
}

// ---------------- async cache->LDS copy (CDNA5 ASYNCcnt path) ----------------
#if defined(__gfx1250__) && __has_builtin(__builtin_amdgcn_global_load_async_to_lds_b128)
#define HAVE_ASYNC_LDS 1
#else
#define HAVE_ASYNC_LDS 0
#endif

__device__ __forceinline__ void async_cp16(void* lds, const void* g) {
#if HAVE_ASYNC_LDS
  __builtin_amdgcn_global_load_async_to_lds_b128(
      (__attribute__((address_space(1))) v4i*)(g),
      (__attribute__((address_space(3))) v4i*)(lds),
      /*offset=*/0, /*cpol=*/0);
#else
  *(U4*)lds = *(const U4*)g;   // global_load_b128 -> ds_store_b128 fallback
#endif
}

__device__ __forceinline__ void wait_async_lds() {
#if HAVE_ASYNC_LDS
#if __has_builtin(__builtin_amdgcn_s_wait_asynccnt)
  __builtin_amdgcn_s_wait_asynccnt(0);
#else
  asm volatile("s_wait_asynccnt 0x0" ::: "memory");
#endif
#endif
}

// ---------------- generic bf16 WMMA GEMM ----------------
// C[M,N] = A[M,K] @ Bm[N,K]^T (+ bias[N]) (+ addend[M,N])
// Optional row permutation for the final logits store:
//   row m -> (m % pB) * pT + (m / pB)
// Requirements: M%32==0, N%128==0, K%32==0, lda/ldb multiples of 8,
// A/Bm 16B aligned.
// Block: 128 threads (4 waves). Block tile 32x128, wave tile 16x64
// (4 WMMA accumulators). Double-buffered LDS staging via async copies.
__global__ __launch_bounds__(128)
void gemm_bf16_wmma(const bf16_t* __restrict__ A, int lda,
                    const bf16_t* __restrict__ Bm, int ldb,
                    const float* __restrict__ bias,
                    const float* __restrict__ addend,
                    float* __restrict__ C,
                    int M, int N, int K,
                    int permute_rows, int pB, int pT) {
  constexpr int TM = 32, TN = 128, TK = 32, LDW = TK + 8;  // +8 bf16 pad (16B)
  __shared__ bf16_t Alds[2][TM][LDW];
  __shared__ bf16_t Blds[2][TN][LDW];

  const int tid  = threadIdx.x;
  const int lane = tid & 31;
  const int wave = tid >> 5;
  const int m0 = blockIdx.x * TM;
  const int n0 = blockIdx.y * TN;
  const int wm = (wave & 1) * 16;   // wave's 16-row slice of the 32-row tile
  const int wn = (wave >> 1) * 64;  // wave's 64-col slice of the 128-col tile

  v8f acc[4] = {v8f{}, v8f{}, v8f{}, v8f{}};

  const int fr = lane & 15;   // fragment row (m or n) within 16
  const int hh = lane >> 4;   // lane half selects K sub-range

  const int arow = tid >> 2;          // 0..31
  const int ach  = (tid & 3) * 8;     // 16B chunk within K tile

  auto stage = [&](int k0, int buf) {
    // A tile 32x32 bf16: one 16B chunk per thread
    async_cp16(&Alds[buf][arow][ach],
               A + (size_t)(m0 + arow) * lda + (k0 + ach));
    // B tile 128x32 bf16: four 16B chunks per thread
#pragma unroll
    for (int r2 = 0; r2 < 4; ++r2) {
      const int c = tid + 128 * r2;
      const int row = c >> 2, ch = (c & 3) * 8;
      async_cp16(&Blds[buf][row][ch],
                 Bm + (size_t)(n0 + row) * ldb + (k0 + ch));
    }
  };

  stage(0, 0);
  const int ksteps = K / TK;
  for (int ks = 0; ks < ksteps; ++ks) {
    const int buf = ks & 1;
    wait_async_lds();   // this wave's pending async copies into `buf` landed
    __syncthreads();    // whole tile resident; all waves done reading buf^1
    if (ks + 1 < ksteps) stage((ks + 1) * TK, buf ^ 1);

    // Fragment layouts per CDNA5 ISA 7.12.2:
    //  A 16x32 bf16: lane l -> M = l&15; elems 0..7: K=8h+e, 8..15: K=16+8h+(e-8)
    //  B 32x16 bf16: lane l -> N = l&15; elems 0..7: K=8h+e, 8..15: K=16+8h+(e-8)
    union Frag { v16bf v; U4 q[2]; };
    Frag fa;
    fa.q[0] = *(const U4*)(&Alds[buf][wm + fr][8 * hh]);
    fa.q[1] = *(const U4*)(&Alds[buf][wm + fr][16 + 8 * hh]);
#pragma unroll
    for (int j = 0; j < 4; ++j) {
      Frag fb;
      fb.q[0] = *(const U4*)(&Blds[buf][wn + 16 * j + fr][8 * hh]);
      fb.q[1] = *(const U4*)(&Blds[buf][wn + 16 * j + fr][16 + 8 * hh]);
      acc[j] = __builtin_amdgcn_wmma_f32_16x16x32_bf16(
          false, fa.v, false, fb.v, (short)0, acc[j], false, false);
    }
  }

  // C layout: lane l -> N = l&15; VGPR v holds M = 8*(l>>4) + v
  const int mbase = m0 + wm + hh * 8;
#pragma unroll
  for (int j = 0; j < 4; ++j) {
    const int cn = n0 + wn + 16 * j + fr;
    const float bv = bias ? bias[cn] : 0.0f;
#pragma unroll
    for (int v = 0; v < 8; ++v) {
      const int m = mbase + v;
      float x = acc[j][v] + bv;
      if (addend) x += addend[(size_t)m * N + cn];
      const size_t ro = permute_rows ? (size_t)((m % pB) * pT + (m / pB))
                                     : (size_t)m;
      C[ro * (size_t)N + cn] = x;
    }
  }
}

// ---------------- fp32 -> bf16 conversion ----------------
__global__ void cvt_bf16_k(const float* __restrict__ in, bf16_t* __restrict__ out,
                           size_t n) {
  size_t i = (size_t)blockIdx.x * blockDim.x + threadIdx.x;
  if (i < n) out[i] = f2bf(in[i]);
}

// transpose-convert: in[R][Ccol] -> out[Ccol][R]
__global__ void cvt_bf16_T_k(const float* __restrict__ in, bf16_t* __restrict__ out,
                             int R, int Ccol) {
  int i = blockIdx.x * blockDim.x + threadIdx.x;
  if (i >= R * Ccol) return;
  int r = i / Ccol, c = i % Ccol;
  out[(size_t)c * R + r] = f2bf(in[i]);
}

__global__ void zero_f32_k(float* __restrict__ p, size_t n) {
  size_t i = (size_t)blockIdx.x * blockDim.x + threadIdx.x;
  if (i < n) p[i] = 0.0f;
}
__global__ void zero_bf16_k(bf16_t* __restrict__ p, size_t n) {
  size_t i = (size_t)blockIdx.x * blockDim.x + threadIdx.x;
  if (i < n) p[i] = __builtin_bit_cast(bf16_t, (unsigned short)0);
}

// out[t*Bdim + b][e] = emb[toks[b*tokStride + t] * Edim + e]   (time-major)
__global__ void embed_gather(const int* __restrict__ toks,
                             const float* __restrict__ emb,
                             bf16_t* __restrict__ out,
                             int Tuse, int Bdim, int Edim, int tokStride) {
  size_t i = (size_t)blockIdx.x * blockDim.x + threadIdx.x;
  size_t total = (size_t)Tuse * Bdim * Edim;
  if (i >= total) return;
  int e  = (int)(i % Edim);
  int tb = (int)(i / Edim);
  int b  = tb % Bdim;
  int t  = tb / Bdim;
  int tok = toks[(size_t)b * tokStride + t];
  out[i] = f2bf(emb[(size_t)tok * Edim + e]);
}

// ---------------- GRU gate pointwise ----------------
// h = (1-z)*tanh(i_n + r*h_n) + z*h_prev ; gates [r|z|n] chunks of H in gi/gh
__global__ void gru_pointwise(const float* __restrict__ gi,
                              const float* __restrict__ gh,
                              const float* __restrict__ hprev,
                              float* __restrict__ hout, bf16_t* __restrict__ hout_bf,
                              float* __restrict__ auxf, int auxfStride,
                              bf16_t* __restrict__ auxb, int auxbStride) {
  int i = blockIdx.x * blockDim.x + threadIdx.x;
  if (i >= B_ * H_) return;
  int b = i / H_, h = i % H_;
  const float* gib = gi + (size_t)b * G_;
  const float* ghb = gh + (size_t)b * G_;
  float r = 1.0f / (1.0f + __expf(-(gib[h] + ghb[h])));
  float z = 1.0f / (1.0f + __expf(-(gib[H_ + h] + ghb[H_ + h])));
  float n = tanhf(gib[2 * H_ + h] + r * ghb[2 * H_ + h]);
  float hn = (1.0f - z) * n + z * hprev[i];
  hout[i] = hn;
  hout_bf[i] = f2bf(hn);
  if (auxf) auxf[(size_t)b * auxfStride + h] = hn;
  if (auxb) auxb[(size_t)b * auxbStride + h] = f2bf(hn);
}

// ---------------- attention: masked softmax + context ----------------
// one block per batch element; q = d1 @ attn_w precomputed (f32 [B,H])
__global__ __launch_bounds__(256)
void attention_step(const float* __restrict__ q, const float* __restrict__ enc,
                    const int* __restrict__ srcTok,
                    float* __restrict__ ctx, bf16_t* __restrict__ ctx_bf,
                    bf16_t* __restrict__ cat_t) {
  __shared__ float wexp[S_];
  __shared__ float sinv;
  const int b = blockIdx.x;
  const int tid = threadIdx.x;
  const float* qb = q + (size_t)b * H_;
  const float* eb = enc + (size_t)b * S_ * H_;

  if (tid < S_) {
    float acc = 0.0f;
    const float* row = eb + (size_t)tid * H_;
    for (int k = 0; k < H_; ++k) acc = fmaf(qb[k], row[k], acc);
    if (srcTok[(size_t)b * S_ + tid] == 0) acc = -1e9f;  // pad mask
    wexp[tid] = acc;
  }
  __syncthreads();
  if (tid == 0) {
    float mx = wexp[0];
    for (int s = 1; s < S_; ++s) mx = fmaxf(mx, wexp[s]);
    float sum = 0.0f;
    for (int s = 0; s < S_; ++s) {
      float e = __expf(wexp[s] - mx);
      wexp[s] = e;
      sum += e;
    }
    sinv = 1.0f / sum;
  }
  __syncthreads();
  const float inv = sinv;
  for (int h = tid; h < H_; h += 256) {
    float acc = 0.0f;
    for (int s = 0; s < S_; ++s) acc = fmaf(wexp[s], eb[(size_t)s * H_ + h], acc);
    acc *= inv;
    ctx[(size_t)b * H_ + h] = acc;
    bf16_t a = f2bf(acc);
    ctx_bf[(size_t)b * H_ + h] = a;
    cat_t[(size_t)b * CH_ + H_ + h] = a;  // concat slot [H:2H)
  }
}

// ---------------- host orchestration ----------------
extern "C" void kernel_launch(void* const* d_in, const int* in_sizes, int n_in,
                              void* d_out, int out_size, void* d_ws, size_t ws_size,
                              hipStream_t stream) {
  (void)in_sizes; (void)n_in; (void)out_size; (void)ws_size;

  const int*   src      = (const int*)d_in[0];
  const int*   tgt      = (const int*)d_in[1];
  const float* enc_emb  = (const float*)d_in[2];
  const float* enc_wih0 = (const float*)d_in[3];
  const float* enc_whh0 = (const float*)d_in[4];
  const float* enc_bih0 = (const float*)d_in[5];
  const float* enc_bhh0 = (const float*)d_in[6];
  const float* enc_wih1 = (const float*)d_in[7];
  const float* enc_whh1 = (const float*)d_in[8];
  const float* enc_bih1 = (const float*)d_in[9];
  const float* enc_bhh1 = (const float*)d_in[10];
  const float* dec_emb  = (const float*)d_in[11];
  const float* dec_wih0 = (const float*)d_in[12];
  const float* dec_whh0 = (const float*)d_in[13];
  const float* dec_bih0 = (const float*)d_in[14];
  const float* dec_bhh0 = (const float*)d_in[15];
  const float* dec_wih1 = (const float*)d_in[16];
  const float* dec_whh1 = (const float*)d_in[17];
  const float* dec_bih1 = (const float*)d_in[18];
  const float* dec_bhh1 = (const float*)d_in[19];
  const float* attn_w   = (const float*)d_in[20];
  const float* out_w    = (const float*)d_in[21];
  const float* out_b    = (const float*)d_in[22];
  float* out = (float*)d_out;

  // workspace bump allocator (256B aligned)
  char* wp = (char*)d_ws;
  auto alloc = [&](size_t bytes) -> void* {
    void* r = (void*)wp;
    wp += (bytes + 255) & ~(size_t)255;
    return r;
  };
  auto abf = [&](size_t n) { return (bf16_t*)alloc(n * sizeof(bf16_t)); };
  auto af  = [&](size_t n) { return (float*)alloc(n * sizeof(float)); };

  bf16_t* w_ewih0 = abf((size_t)G_ * E_);
  bf16_t* w_ewhh0 = abf((size_t)G_ * H_);
  bf16_t* w_ewih1 = abf((size_t)G_ * H_);
  bf16_t* w_ewhh1 = abf((size_t)G_ * H_);
  bf16_t* w_dwih0 = abf((size_t)G_ * (E_ + H_));
  bf16_t* w_dwhh0 = abf((size_t)G_ * H_);
  bf16_t* w_dwih1 = abf((size_t)G_ * H_);
  bf16_t* w_dwhh1 = abf((size_t)G_ * H_);
  bf16_t* w_attnT = abf((size_t)H_ * H_);
  bf16_t* w_out   = abf((size_t)VT_ * CH_);
  bf16_t* xenc    = abf((size_t)S_ * B_ * E_);
  bf16_t* xdec    = abf((size_t)Tm1_ * B_ * E_);
  bf16_t* h0b     = abf((size_t)B_ * H_);
  bf16_t* h1b     = abf((size_t)B_ * H_);
  bf16_t* ctxb    = abf((size_t)B_ * H_);
  bf16_t* catb    = abf((size_t)Tm1_ * B_ * CH_);
  float* gi0e = af((size_t)S_ * B_ * G_);
  float* gi0d = af((size_t)Tm1_ * B_ * G_);
  float* gha  = af((size_t)B_ * G_);
  float* gi1s = af((size_t)B_ * G_);
  float* gh1s = af((size_t)B_ * G_);
  float* gi0s = af((size_t)B_ * G_);
  float* h0   = af((size_t)B_ * H_);
  float* h1   = af((size_t)B_ * H_);
  float* ctx  = af((size_t)B_ * H_);
  float* qbuf = af((size_t)B_ * H_);
  float* enco = af((size_t)B_ * S_ * H_);

  auto cvt = [&](const float* i, bf16_t* o, size_t n) {
    cvt_bf16_k<<<dim3((unsigned)((n + 255) / 256)), dim3(256), 0, stream>>>(i, o, n);
  };
  // weights -> bf16 (L2-resident working set, reused across all 127 steps)
  cvt(enc_wih0, w_ewih0, (size_t)G_ * E_);
  cvt(enc_whh0, w_ewhh0, (size_t)G_ * H_);
  cvt(enc_wih1, w_ewih1, (size_t)G_ * H_);
  cvt(enc_whh1, w_ewhh1, (size_t)G_ * H_);
  cvt(dec_wih0, w_dwih0, (size_t)G_ * (E_ + H_));
  cvt(dec_whh0, w_dwhh0, (size_t)G_ * H_);
  cvt(dec_wih1, w_dwih1, (size_t)G_ * H_);
  cvt(dec_whh1, w_dwhh1, (size_t)G_ * H_);
  cvt(out_w,    w_out,   (size_t)VT_ * CH_);
  cvt_bf16_T_k<<<dim3((H_ * H_ + 255) / 256), dim3(256), 0, stream>>>(
      attn_w, w_attnT, H_, H_);  // q GEMM needs attn_w as [h][k]

  // embeddings (time-major, bf16)
  embed_gather<<<dim3((S_ * B_ * E_ + 255) / 256), dim3(256), 0, stream>>>(
      src, enc_emb, xenc, S_, B_, E_, S_);
  embed_gather<<<dim3((Tm1_ * B_ * E_ + 255) / 256), dim3(256), 0, stream>>>(
      tgt, dec_emb, xdec, Tm1_, B_, E_, T_);

  auto zf = [&](float* p, size_t n) {
    zero_f32_k<<<dim3((unsigned)((n + 255) / 256)), dim3(256), 0, stream>>>(p, n);
  };
  auto zb = [&](bf16_t* p, size_t n) {
    zero_bf16_k<<<dim3((unsigned)((n + 255) / 256)), dim3(256), 0, stream>>>(p, n);
  };
  zf(h0, (size_t)B_ * H_);  zf(h1, (size_t)B_ * H_);  zf(ctx, (size_t)B_ * H_);
  zb(h0b, (size_t)B_ * H_); zb(h1b, (size_t)B_ * H_); zb(ctxb, (size_t)B_ * H_);

  auto gemm = [&](const bf16_t* A, int lda, const bf16_t* Bm, int ldb,
                  const float* bias, const float* addend, float* C,
                  int M, int N, int K, int permute, int pb, int pt) {
    gemm_bf16_wmma<<<dim3(M / 32, N / 128), dim3(128), 0, stream>>>(
        A, lda, Bm, ldb, bias, addend, C, M, N, K, permute, pb, pt);
  };
  auto gru = [&](const float* gi, const float* gh, float* h, bf16_t* hb,
                 float* auxf, int auxfs, bf16_t* auxb, int auxbs) {
    gru_pointwise<<<dim3(B_ * H_ / 256), dim3(256), 0, stream>>>(
        gi, gh, h, h, hb, auxf, auxfs, auxb, auxbs);
  };

  // ---- encoder ----
  // hoisted input GEMM over all timesteps: [2048,256] @ [1536,256]^T
  gemm(xenc, E_, w_ewih0, E_, enc_bih0, nullptr, gi0e, S_ * B_, G_, E_, 0, 0, 0);
  for (int t = 0; t < S_; ++t) {
    gemm(h0b, H_, w_ewhh0, H_, enc_bhh0, nullptr, gha, B_, G_, H_, 0, 0, 0);
    gru(gi0e + (size_t)t * B_ * G_, gha, h0, h0b, nullptr, 0, nullptr, 0);
    gemm(h0b, H_, w_ewih1, H_, enc_bih1, nullptr, gi1s, B_, G_, H_, 0, 0, 0);
    gemm(h1b, H_, w_ewhh1, H_, enc_bhh1, nullptr, gh1s, B_, G_, H_, 0, 0, 0);
    gru(gi1s, gh1s, h1, h1b, enco + (size_t)t * H_, S_ * H_, nullptr, 0);
  }

  // ---- decoder (teacher forcing); initial state = encoder final h0/h1 ----
  // hoisted embedding part of layer-0 input GEMM (first E columns of dec_wih0)
  gemm(xdec, E_, w_dwih0, E_ + H_, nullptr, nullptr, gi0d, Tm1_ * B_, G_, E_, 0, 0, 0);
  for (int t = 0; t < Tm1_; ++t) {
    // ctx part of layer-0 input (columns E..E+H of dec_wih0) + bias + emb part
    gemm(ctxb, H_, w_dwih0 + E_, E_ + H_, dec_bih0,
         gi0d + (size_t)t * B_ * G_, gi0s, B_, G_, H_, 0, 0, 0);
    gemm(h0b, H_, w_dwhh0, H_, dec_bhh0, nullptr, gha, B_, G_, H_, 0, 0, 0);
    gru(gi0s, gha, h0, h0b, nullptr, 0, nullptr, 0);
    gemm(h0b, H_, w_dwih1, H_, dec_bih1, nullptr, gi1s, B_, G_, H_, 0, 0, 0);
    gemm(h1b, H_, w_dwhh1, H_, dec_bhh1, nullptr, gh1s, B_, G_, H_, 0, 0, 0);
    // d1 -> state + concat slot [0:H) of cat[t]
    gru(gi1s, gh1s, h1, h1b, nullptr, 0, catb + (size_t)t * B_ * CH_, CH_);
    // q = d1 @ attn_w   ([32,512] @ [512,512]^T with transposed weight)
    gemm(h1b, H_, w_attnT, H_, nullptr, nullptr, qbuf, B_, H_, H_, 0, 0, 0);
    attention_step<<<dim3(B_), dim3(256), 0, stream>>>(
        qbuf, enco, src, ctx, ctxb, catb + (size_t)t * B_ * CH_);
  }

  // ---- batched output projection (dominant GEMM): logits -> d_out[B,T-1,VT]
  gemm(catb, CH_, w_out, CH_, out_b, nullptr, out,
       Tm1_ * B_, VT_, CH_, /*permute=*/1, B_, Tm1_);
}
